// Diffusion_55800215110127
// MI455X (gfx1250) — matrix operand order repaired and checked by reading, not compiled, not run on hardware
//
#include <hip/hip_runtime.h>
#include <hip/hip_bf16.h>
#include <math.h>

// ---------------------------------------------------------------------------
// Types for CDNA5 WMMA (gfx1250, wave32)
// ---------------------------------------------------------------------------
typedef __bf16 bf16_t;
typedef __bf16 v16bf __attribute__((ext_vector_type(16)));
typedef __bf16 v8bf  __attribute__((ext_vector_type(8)));
typedef __bf16 v4bf  __attribute__((ext_vector_type(4)));
typedef float  v8f   __attribute__((ext_vector_type(8)));

__device__ __forceinline__ float warp_sum(float v) {
#pragma unroll
  for (int o = 16; o > 0; o >>= 1) v += __shfl_xor(v, o, 32);
  return v;
}
// sum within 8-lane segments (lanes grouped by lane>>3)
__device__ __forceinline__ float seg8_sum(float v) {
  v += __shfl_xor(v, 1, 32);
  v += __shfl_xor(v, 2, 32);
  v += __shfl_xor(v, 4, 32);
  return v;
}

// monotone float -> uint key for atomic max (all keys >= 0; 0 ~ -inf sentinel)
__device__ __forceinline__ unsigned fkey(float f) {
  unsigned b = __float_as_uint(f);
  return (b & 0x80000000u) ? ~b : (b | 0x80000000u);
}
__device__ __forceinline__ float fval(unsigned k) {
  unsigned b = (k & 0x80000000u) ? (k ^ 0x80000000u) : ~k;
  return __uint_as_float(b);
}

// ---------------------------------------------------------------------------
// 0. beta-schedule tables (module constants in the reference; fp64 cumprod)
// ---------------------------------------------------------------------------
__global__ void build_tables_k(float* __restrict__ sab, float* __restrict__ s1ab) {
  if (threadIdx.x == 0 && blockIdx.x == 0) {
    double ab = 1.0;
    for (int i = 0; i < 1000; ++i) {
      double beta = 1e-4 + (0.02 - 1e-4) * (double)i / 999.0;
      ab *= (1.0 - beta);
      sab[i]  = (float)sqrt(ab);
      s1ab[i] = (float)sqrt(1.0 - ab);
    }
  }
}

// ---------------------------------------------------------------------------
// 1. fp32 -> bf16 cast / fused sum->bf16 (vectorized, n % 4 == 0 always here)
// ---------------------------------------------------------------------------
__global__ __launch_bounds__(256) void cast_f32_bf16_k(const float* __restrict__ in,
                                                       bf16_t* __restrict__ out, int n4) {
  for (int i = blockIdx.x * 256 + threadIdx.x; i < n4; i += gridDim.x * 256) {
    float4 v = ((const float4*)in)[i];
    v4bf o = {(bf16_t)v.x, (bf16_t)v.y, (bf16_t)v.z, (bf16_t)v.w};
    ((v4bf*)out)[i] = o;
  }
}

__global__ __launch_bounds__(256) void sum_to_bf16_k(const float* __restrict__ a,
                                                     const float* __restrict__ b,
                                                     bf16_t* __restrict__ out, int n4) {
  for (int i = blockIdx.x * 256 + threadIdx.x; i < n4; i += gridDim.x * 256) {
    float4 u = ((const float4*)a)[i];
    float4 v = ((const float4*)b)[i];
    v4bf o = {(bf16_t)(u.x + v.x), (bf16_t)(u.y + v.y), (bf16_t)(u.z + v.z),
              (bf16_t)(u.w + v.w)};
    ((v4bf*)out)[i] = o;
  }
}

// ---------------------------------------------------------------------------
// 2. LayerNorm(x) + q-sample, one wave per 128-wide row -> bf16 x_t
// ---------------------------------------------------------------------------
__global__ __launch_bounds__(256) void ln_qsample_k(
    const float* __restrict__ x, const float* __restrict__ noise,
    const int* __restrict__ t, const float* __restrict__ sab,
    const float* __restrict__ s1ab, bf16_t* __restrict__ out, int N) {
  int n = blockIdx.x * 8 + (threadIdx.x >> 5);
  if (n >= N) return;
  int lane = threadIdx.x & 31;
  float4 v = ((const float4*)(x + (size_t)n * 128))[lane];
  float s = warp_sum(v.x + v.y + v.z + v.w);
  float mean = s * (1.f / 128.f);
  float dx = v.x - mean, dy = v.y - mean, dz = v.z - mean, dw = v.w - mean;
  float vs = warp_sum(dx * dx + dy * dy + dz * dz + dw * dw);
  float rstd = rsqrtf(vs * (1.f / 128.f) + 1e-5f);
  int ti = t[n];
  float a = sab[ti], b = s1ab[ti];
  float4 nv = ((const float4*)(noise + (size_t)n * 128))[lane];
  v4bf o = {(bf16_t)(a * (dx * rstd) + b * nv.x),
            (bf16_t)(a * (dy * rstd) + b * nv.y),
            (bf16_t)(a * (dz * rstd) + b * nv.z),
            (bf16_t)(a * (dw * rstd) + b * nv.w)};
  ((v4bf*)(out + (size_t)n * 128))[lane] = o;
}

// ---------------------------------------------------------------------------
// 3. LN(+affine)+ReLU -> bf16, one wave per W-wide row
// ---------------------------------------------------------------------------
template <int W>
__global__ __launch_bounds__(256) void ln_relu_bf16_k(
    const float* __restrict__ in, const float* __restrict__ g,
    const float* __restrict__ be, bf16_t* __restrict__ out, int N) {
  int n = blockIdx.x * 8 + (threadIdx.x >> 5);
  if (n >= N) return;
  int lane = threadIdx.x & 31;
  constexpr int PER = W / 128;  // float4 chunks per lane (1 or 2)
  const float4* row = (const float4*)(in + (size_t)n * W);
  float4 v[PER];
  float s = 0.f;
#pragma unroll
  for (int i = 0; i < PER; ++i) {
    v[i] = row[lane + 32 * i];
    s += v[i].x + v[i].y + v[i].z + v[i].w;
  }
  s = warp_sum(s);
  float mean = s * (1.f / (float)W);
  float vs = 0.f;
#pragma unroll
  for (int i = 0; i < PER; ++i) {
    float a = v[i].x - mean, b = v[i].y - mean, c = v[i].z - mean, d = v[i].w - mean;
    vs += a * a + b * b + c * c + d * d;
  }
  vs = warp_sum(vs);
  float rstd = rsqrtf(vs * (1.f / (float)W) + 1e-5f);
  const float4* gp = (const float4*)g;
  const float4* bp = (const float4*)be;
  v4bf* orow = (v4bf*)(out + (size_t)n * W);
#pragma unroll
  for (int i = 0; i < PER; ++i) {
    int c4 = lane + 32 * i;
    float4 gg = gp[c4], bb = bp[c4];
    v4bf o = {(bf16_t)fmaxf((v[i].x - mean) * rstd * gg.x + bb.x, 0.f),
              (bf16_t)fmaxf((v[i].y - mean) * rstd * gg.y + bb.y, 0.f),
              (bf16_t)fmaxf((v[i].z - mean) * rstd * gg.z + bb.z, 0.f),
              (bf16_t)fmaxf((v[i].w - mean) * rstd * gg.w + bb.w, 0.f)};
    orow[c4] = o;
  }
}

// ---------------------------------------------------------------------------
// 4. WMMA bf16 GEMM: C[M,O] = A[M,K] @ B[K,O] + bias (+resid) (+ReLU)
//    block = 256 threads (8 wave32), 64x128 C tile, each wave a 32x32 register
//    tile (4 accumulators -> 2 fragment loads per WMMA). K chunked by 64
//    through LDS. LDS stride 72 (=144B) keeps every v8bf fragment read 16B-
//    aligned and bank-conflict-free (row*36 dwords mod 64 tiles all banks).
// ---------------------------------------------------------------------------
template <bool RESID, bool RELU>
__global__ __launch_bounds__(256) void gemm_bf16_wmma_k(
    const bf16_t* __restrict__ A, int lda,
    const bf16_t* __restrict__ B, int ldb,
    const float* __restrict__ bias,
    const float* __restrict__ resid,
    float* __restrict__ outF, bf16_t* __restrict__ outB,
    int M, int K, int O) {
  constexpr int LDP = 72;
  __shared__ __align__(32) bf16_t As[64 * LDP];   // [row][k]
  __shared__ __align__(32) bf16_t Bs[128 * LDP];  // [col][k] (transposed stage)

  const int tid  = threadIdx.x;
  const int wave = tid >> 5;
  const int lane = tid & 31;
  const int lr   = lane & 15;
  const int hi   = lane >> 4;

  const int row0 = blockIdx.x * 64;
  const int col0 = blockIdx.y * 128;
  const int wr   = (wave & 1) * 32;   // local wave row (2 tiles of 16)
  const int wc   = (wave >> 1) * 32;  // local wave col (2 tiles of 16)

  v8f acc[2][2] = {};

  for (int k0 = 0; k0 < K; k0 += 64) {
    // ---- stage A: 64 rows x 64 k, 16B vector loads, per-row guard ----
#pragma unroll
    for (int i = 0; i < 2; ++i) {
      int idx = tid + i * 256;      // 0..511
      int r   = idx >> 3;           // row 0..63
      int c8  = (idx & 7) * 8;      // k offset 0..56
      int row = row0 + r;
      v8bf v = {};
      if (row < M) {
        const bf16_t* gp = &A[(size_t)row * lda + k0 + c8];
        v = *(const v8bf*)gp;
        if (k0 + 64 < K) __builtin_prefetch(gp + 64, 0, 1);
      }
      *(v8bf*)&As[r * LDP + c8] = v;
    }
    // ---- stage B transposed: 64 k x 128 col, coalesced 16B loads ----
#pragma unroll
    for (int i = 0; i < 4; ++i) {
      int idx = tid + i * 256;      // 0..1023
      int kr  = idx >> 4;           // k row 0..63
      int c8  = (idx & 15) * 8;     // col 0..120
      v8bf v = *(const v8bf*)&B[(size_t)(k0 + kr) * ldb + col0 + c8];
#pragma unroll
      for (int q = 0; q < 8; ++q) Bs[(c8 + q) * LDP + kr] = v[q];
    }
    __syncthreads();

#pragma unroll
    for (int kk = 0; kk < 64; kk += 32) {
      // A fragment 16x32 bf16 (this code's contiguous k order, same as B):
      //  lanes 0-15: elems0-15 = K kk..+15
      //  lanes16-31: elems0-15 = K kk+16..+31
      v16bf af[2];
#pragma unroll
      for (int rs = 0; rs < 2; ++rs) {
        const bf16_t* ap = &As[(wr + rs * 16 + lr) * LDP + kk + hi * 16];
        v8bf lo = *(const v8bf*)ap;
        v8bf hi8 = *(const v8bf*)(ap + 8);
        af[rs] = __builtin_shufflevector(lo, hi8, 0, 1, 2, 3, 4, 5, 6, 7, 8, 9,
                                         10, 11, 12, 13, 14, 15);
      }
      // B fragment 32x16 bf16: lanes 0-15 K kk..+15 (N=lr); lanes16-31 K kk+16..+31
      v16bf bf[2];
#pragma unroll
      for (int cs = 0; cs < 2; ++cs) {
        const bf16_t* bp = &Bs[(wc + cs * 16 + lr) * LDP + kk + hi * 16];
        v8bf lo = *(const v8bf*)bp;
        v8bf hi8 = *(const v8bf*)(bp + 8);
        bf[cs] = __builtin_shufflevector(lo, hi8, 0, 1, 2, 3, 4, 5, 6, 7, 8, 9,
                                         10, 11, 12, 13, 14, 15);
      }
#pragma unroll
      for (int rs = 0; rs < 2; ++rs)
#pragma unroll
        for (int cs = 0; cs < 2; ++cs)
          acc[rs][cs] = __builtin_amdgcn_wmma_f32_16x16x32_bf16(
              false, af[rs], false, bf[cs], (short)0, acc[rs][cs], false, false);
    }
    __syncthreads();
  }

  // C/D layout: VGPR v -> M = v (lanes 0-15) / v+8 (lanes 16-31), N = lr
#pragma unroll
  for (int rs = 0; rs < 2; ++rs) {
#pragma unroll
    for (int cs = 0; cs < 2; ++cs) {
      int col = col0 + wc + cs * 16 + lr;
      float bb = bias[col];
#pragma unroll
      for (int v = 0; v < 8; ++v) {
        int row = row0 + wr + rs * 16 + hi * 8 + v;
        if (row < M) {
          size_t ix = (size_t)row * O + col;
          float xa = acc[rs][cs][v] + bb;
          if (RESID) xa += resid[ix];
          if (RELU) xa = fmaxf(xa, 0.f);
          if (outF) outF[ix] = xa;
          if (outB) outB[ix] = (bf16_t)xa;
        }
      }
    }
  }
}

// ---------------------------------------------------------------------------
// 5. GATv2 edge pass 1: logits + segment-max (atomic uint-key max), wave/edge
//    float4 gathers; 4-head dot reduced with 3 segmented shuffles.
// ---------------------------------------------------------------------------
template <int HEADS>
__global__ __launch_bounds__(256) void edge_logits_k(
    const float* __restrict__ fs, const float* __restrict__ fd,
    const int* __restrict__ src, const int* __restrict__ dst,
    const float* __restrict__ attn, float* __restrict__ logits,
    unsigned* __restrict__ mkey, int E) {
  int e = blockIdx.x * 8 + (threadIdx.x >> 5);
  if (e >= E) return;
  int lane = threadIdx.x & 31;
  int s = src[e], d = dst[e];
  float4 a  = ((const float4*)(fs + (size_t)s * 128))[lane];
  float4 b  = ((const float4*)(fd + (size_t)d * 128))[lane];
  float4 at = ((const float4*)attn)[lane];
  float vx = a.x + b.x; vx = (vx > 0.f) ? vx : 0.2f * vx;
  float vy = a.y + b.y; vy = (vy > 0.f) ? vy : 0.2f * vy;
  float vz = a.z + b.z; vz = (vz > 0.f) ? vz : 0.2f * vz;
  float vw = a.w + b.w; vw = (vw > 0.f) ? vw : 0.2f * vw;
  float p = vx * at.x + vy * at.y + vz * at.z + vw * at.w;
  if (HEADS == 4) {
    // lane covers j = lane*4..lane*4+3, all in head = lane>>3 (Dh = 32)
    p = seg8_sum(p);
    if ((lane & 7) == 0) {
      int h = lane >> 3;
      logits[(size_t)e * 4 + h] = p;
      atomicMax(&mkey[(size_t)d * 4 + h], fkey(p));
    }
  } else {
    p = warp_sum(p);
    if (lane == 0) {
      logits[e] = p;
      atomicMax(&mkey[d], fkey(p));
    }
  }
}

// decode segment-max keys; key==0 (no edges) or non-finite -> 0 (ref semantics)
__global__ __launch_bounds__(256) void mfin_k(const unsigned* __restrict__ mkey,
                                              float* __restrict__ m, int nh) {
  int i = blockIdx.x * 256 + threadIdx.x;
  if (i >= nh) return;
  unsigned k = mkey[i];
  float v = 0.f;
  if (k != 0u) {
    float f = fval(k);
    unsigned b = __float_as_uint(f);
    if (((b >> 23) & 0xFFu) != 0xFFu) v = f;  // finite check
  }
  m[i] = v;
}

// ---------------------------------------------------------------------------
// 6. GATv2 edge pass 2: w = exp(logit - m[dst]); den += w; out += fs[src]*w
// ---------------------------------------------------------------------------
template <int HEADS>
__global__ __launch_bounds__(256) void edge_accum_k(
    const float* __restrict__ fs, const int* __restrict__ src,
    const int* __restrict__ dst, const float* __restrict__ logits,
    const float* __restrict__ m, float* __restrict__ den,
    float* __restrict__ gout, int E) {
  int e = blockIdx.x * 8 + (threadIdx.x >> 5);
  if (e >= E) return;
  int lane = threadIdx.x & 31;
  int s = src[e], d = dst[e];
  float w = 0.f;
  if (lane < HEADS) {
    w = expf(logits[(size_t)e * HEADS + lane] - m[(size_t)d * HEADS + lane]);
    atomicAdd(&den[(size_t)d * HEADS + lane], w);
  }
  float wj = __shfl(w, (HEADS == 4) ? (lane >> 3) : 0, 32);
  float4 fv = ((const float4*)(fs + (size_t)s * 128))[lane];
  float* gop = gout + (size_t)d * 128 + lane * 4;
  atomicAdd(gop + 0, fv.x * wj);
  atomicAdd(gop + 1, fv.y * wj);
  atomicAdd(gop + 2, fv.z * wj);
  atomicAdd(gop + 3, fv.w * wj);
}

// out[n,j] = acc[n,j] / max(den[n, j/Dh], 1e-9); dual f32/bf16 store
template <int HEADS>
__global__ __launch_bounds__(256) void gat_norm_k(
    const float* __restrict__ acc, const float* __restrict__ den,
    float* __restrict__ outF, bf16_t* __restrict__ outB, int N) {
  int total4 = N * 32;  // float4 groups
  for (int i = blockIdx.x * 256 + threadIdx.x; i < total4; i += gridDim.x * 256) {
    int n = i >> 5, j = (i & 31) * 4;  // j..j+3 share a head (Dh >= 32)
    constexpr int DH = 128 / HEADS;
    float rden = 1.f / fmaxf(den[(size_t)n * HEADS + j / DH], 1e-9f);
    float4 v = ((const float4*)acc)[i];
    v.x *= rden; v.y *= rden; v.z *= rden; v.w *= rden;
    if (outF) ((float4*)outF)[i] = v;
    if (outB) {
      v4bf o = {(bf16_t)v.x, (bf16_t)v.y, (bf16_t)v.z, (bf16_t)v.w};
      ((v4bf*)outB)[i] = o;
    }
  }
}

// ---------------------------------------------------------------------------
// 7. cosine loss: mean((1 - cos(out, y))^2), wave per node
// ---------------------------------------------------------------------------
__global__ __launch_bounds__(256) void loss_k(const float* __restrict__ out,
                                              const float* __restrict__ y,
                                              float* __restrict__ loss, int N) {
  int n = blockIdx.x * 8 + (threadIdx.x >> 5);
  if (n >= N) return;
  int lane = threadIdx.x & 31;
  float4 a = ((const float4*)(out + (size_t)n * 128))[lane];
  float4 b = ((const float4*)(y + (size_t)n * 128))[lane];
  float dd = warp_sum(a.x * b.x + a.y * b.y + a.z * b.z + a.w * b.w);
  float no = warp_sum(a.x * a.x + a.y * a.y + a.z * a.z + a.w * a.w);
  float ny = warp_sum(b.x * b.x + b.y * b.y + b.z * b.z + b.w * b.w);
  if (lane == 0) {
    float c = dd / (fmaxf(sqrtf(no), 1e-12f) * fmaxf(sqrtf(ny), 1e-12f));
    float t = 1.f - c;
    atomicAdd(loss, t * t / (float)N);
  }
}

// ---------------------------------------------------------------------------
// Host orchestration
// ---------------------------------------------------------------------------
namespace {
struct Carver {
  char* p; size_t off;
  void* take(size_t bytes) {
    size_t a = (off + 255) & ~(size_t)255;
    void* r = p + a;
    off = a + bytes;
    return r;
  }
};
struct MlpP { const float *W1, *W2, *Wi, *Wo, *b1, *b2, *be, *bi, *bo, *g; };
}  // namespace

extern "C" void kernel_launch(void* const* d_in, const int* in_sizes, int n_in,
                              void* d_out, int out_size, void* d_ws, size_t ws_size,
                              hipStream_t stream) {
  (void)n_in; (void)out_size; (void)ws_size;
  // ----- inputs (JAX pytree flatten: sorted dict keys, lists in order) -----
  const float* x     = (const float*)d_in[0];
  const float* y     = (const float*)d_in[1];
  const float* noise = (const float*)d_in[2];
  // params: down[2]{Wd,Ws,attn,bd,bs}, mlp_in, mlp_mid, mlp_out (W1,W2,Wi,Wo,b1,b2,be,bi,bo,g),
  //         time_emb, up[2]{Wd,Ws,attn,bd,bs}
  const float* dWd[2] = {(const float*)d_in[3], (const float*)d_in[8]};
  const float* dWs[2] = {(const float*)d_in[4], (const float*)d_in[9]};
  const float* dAt[2] = {(const float*)d_in[5], (const float*)d_in[10]};
  const float* dBd[2] = {(const float*)d_in[6], (const float*)d_in[11]};
  const float* dBs[2] = {(const float*)d_in[7], (const float*)d_in[12]};
  auto mlpAt = [&](int b) {
    MlpP m;
    m.W1 = (const float*)d_in[b + 0]; m.W2 = (const float*)d_in[b + 1];
    m.Wi = (const float*)d_in[b + 2]; m.Wo = (const float*)d_in[b + 3];
    m.b1 = (const float*)d_in[b + 4]; m.b2 = (const float*)d_in[b + 5];
    m.be = (const float*)d_in[b + 6]; m.bi = (const float*)d_in[b + 7];
    m.bo = (const float*)d_in[b + 8]; m.g  = (const float*)d_in[b + 9];
    return m;
  };
  MlpP mIn = mlpAt(13), mMid = mlpAt(23), mOut = mlpAt(33);
  const float* uWd[2] = {(const float*)d_in[44], (const float*)d_in[49]};
  const float* uWs[2] = {(const float*)d_in[45], (const float*)d_in[50]};
  const float* uAt[2] = {(const float*)d_in[46], (const float*)d_in[51]};
  const float* uBd[2] = {(const float*)d_in[47], (const float*)d_in[52]};
  const float* uBs[2] = {(const float*)d_in[48], (const float*)d_in[53]};
  const int* t   = (const int*)d_in[54];
  const int* src = (const int*)d_in[55];
  const int* dst = (const int*)d_in[56];

  const int N = in_sizes[0] / 128;
  const int E = in_sizes[55];

  // ----- workspace carve -----
  Carver ws{(char*)d_ws, 0};
  float*  SAB  = (float*)ws.take(1000 * 4);
  float*  S1AB = (float*)ws.take(1000 * 4);
  bf16_t* WARENA = (bf16_t*)ws.take((size_t)458752 * 2);
  bf16_t* XT   = (bf16_t*)ws.take((size_t)N * 128 * 2);
  bf16_t* BF_H = (bf16_t*)ws.take((size_t)N * 128 * 2);
  bf16_t* BF_A = (bf16_t*)ws.take((size_t)N * 256 * 2);
  bf16_t* BF_R = (bf16_t*)ws.take((size_t)N * 256 * 2);
  float*  F_H  = (float*)ws.take((size_t)N * 256 * 4);
  float*  F_T  = (float*)ws.take((size_t)N * 256 * 4);
  float*  F_G1 = (float*)ws.take((size_t)N * 128 * 4);
  float*  F_G2 = (float*)ws.take((size_t)N * 128 * 4);
  float*  F_GO = (float*)ws.take((size_t)N * 128 * 4);
  float*  F_SK0 = (float*)ws.take((size_t)N * 128 * 4);
  float*  F_SK1 = (float*)ws.take((size_t)N * 128 * 4);
  float*  F_HC = (float*)ws.take((size_t)N * 128 * 4);
  float*  LOG  = (float*)ws.take((size_t)E * 4 * 4);
  unsigned* MKEY = (unsigned*)ws.take((size_t)N * 4 * 4);
  float*  MVAL = (float*)ws.take((size_t)N * 4 * 4);
  float*  DEN  = (float*)ws.take((size_t)N * 4 * 4);

  // ----- weight casts to bf16 arena -----
  size_t woff = 0;
  auto castW = [&](const float* w, int n) {
    bf16_t* p = WARENA + woff; woff += (size_t)n;
    int n4 = n / 4;
    cast_f32_bf16_k<<<(n4 + 255) / 256, 256, 0, stream>>>(w, p, n4);
    return p;
  };
  bf16_t* inWi = castW(mIn.Wi, 128 * 256);
  bf16_t* inW1 = castW(mIn.W1, 256 * 256);
  bf16_t* inW2 = castW(mIn.W2, 256 * 256);
  bf16_t* inWo = castW(mIn.Wo, 256 * 128);
  bf16_t* miWi = castW(mMid.Wi, 128 * 128);
  bf16_t* miW1 = castW(mMid.W1, 128 * 128);
  bf16_t* miW2 = castW(mMid.W2, 128 * 128);
  bf16_t* miWo = castW(mMid.Wo, 128 * 128);
  bf16_t* moWi = castW(mOut.Wi, 128 * 128);
  bf16_t* moW1 = castW(mOut.W1, 128 * 128);
  bf16_t* moW2 = castW(mOut.W2, 128 * 128);
  bf16_t* moWo = castW(mOut.Wo, 128 * 128);
  bf16_t *gWs[4], *gWd[4];
  for (int l = 0; l < 2; ++l) {
    gWs[l] = castW(dWs[l], 128 * 128);
    gWd[l] = castW(dWd[l], 128 * 128);
  }
  for (int l = 0; l < 2; ++l) {
    gWs[2 + l] = castW(uWs[l], 128 * 128);
    gWd[2 + l] = castW(uWd[l], 128 * 128);
  }

  build_tables_k<<<1, 1, 0, stream>>>(SAB, S1AB);

  // ----- GEMM dispatch (64x128 block tile) -----
  auto gemm = [&](const bf16_t* A, int lda, const bf16_t* B, int ldb,
                  const float* bias, const float* resid, float* outF,
                  bf16_t* outB, int M, int K, int O, bool relu) {
    dim3 g((M + 63) / 64, O / 128), b(256);
    if (resid) {
      if (relu) gemm_bf16_wmma_k<true, true><<<g, b, 0, stream>>>(A, lda, B, ldb, bias, resid, outF, outB, M, K, O);
      else      gemm_bf16_wmma_k<true, false><<<g, b, 0, stream>>>(A, lda, B, ldb, bias, resid, outF, outB, M, K, O);
    } else {
      if (relu) gemm_bf16_wmma_k<false, true><<<g, b, 0, stream>>>(A, lda, B, ldb, bias, resid, outF, outB, M, K, O);
      else      gemm_bf16_wmma_k<false, false><<<g, b, 0, stream>>>(A, lda, B, ldb, bias, resid, outF, outB, M, K, O);
    }
  };

  // ----- MLP: h=in@Wi+bi; r=relu(ln(h@W1+b1)*g+be); h+=r@W2+b2; out=relu(h@Wo+bo)
  auto run_mlp = [&](const MlpP& m, bf16_t* Wi, bf16_t* W1, bf16_t* W2,
                     bf16_t* Wo, int din, int dh, int dout, const bf16_t* inbf,
                     float* outF, bf16_t* outB) {
    gemm(inbf, din, Wi, dh, m.bi, nullptr, F_H, BF_R, N, din, dh, false);
    gemm(BF_R, dh, W1, dh, m.b1, nullptr, F_T, nullptr, N, dh, dh, false);
    if (dh == 256)
      ln_relu_bf16_k<256><<<(N + 7) / 8, 256, 0, stream>>>(F_T, m.g, m.be, BF_A, N);
    else
      ln_relu_bf16_k<128><<<(N + 7) / 8, 256, 0, stream>>>(F_T, m.g, m.be, BF_A, N);
    gemm(BF_A, dh, W2, dh, m.b2, F_H, nullptr, BF_R, N, dh, dh, false);
    gemm(BF_R, dh, Wo, dout, m.bo, nullptr, outF, outB, N, dh, dout, true);
  };

  // ----- GATv2 layer -----
  auto run_gat = [&](int heads, const bf16_t* inbf, const bf16_t* Ws,
                     const float* bs, const bf16_t* Wd, const float* bd,
                     const float* attn, float* outF, bf16_t* outB) {
    gemm(inbf, 128, Ws, 128, bs, nullptr, F_G1, nullptr, N, 128, 128, false);
    gemm(inbf, 128, Wd, 128, bd, nullptr, F_G2, nullptr, N, 128, 128, false);
    hipMemsetAsync(MKEY, 0, (size_t)N * heads * 4, stream);
    dim3 ge((E + 7) / 8), be(256);
    if (heads == 4)
      edge_logits_k<4><<<ge, be, 0, stream>>>(F_G1, F_G2, src, dst, attn, LOG, MKEY, E);
    else
      edge_logits_k<1><<<ge, be, 0, stream>>>(F_G1, F_G2, src, dst, attn, LOG, MKEY, E);
    mfin_k<<<(N * heads + 255) / 256, 256, 0, stream>>>(MKEY, MVAL, N * heads);
    hipMemsetAsync(DEN, 0, (size_t)N * heads * 4, stream);
    hipMemsetAsync(F_GO, 0, (size_t)N * 128 * 4, stream);
    if (heads == 4)
      edge_accum_k<4><<<ge, be, 0, stream>>>(F_G1, src, dst, LOG, MVAL, DEN, F_GO, E);
    else
      edge_accum_k<1><<<ge, be, 0, stream>>>(F_G1, src, dst, LOG, MVAL, DEN, F_GO, E);
    int blks = ((N * 32) + 255) / 256;
    if (heads == 4)
      gat_norm_k<4><<<blks, 256, 0, stream>>>(F_GO, DEN, outF, outB, N);
    else
      gat_norm_k<1><<<blks, 256, 0, stream>>>(F_GO, DEN, outF, outB, N);
  };

  // ===== pipeline =====
  ln_qsample_k<<<(N + 7) / 8, 256, 0, stream>>>(x, noise, t, SAB, S1AB, XT, N);

  run_mlp(mIn, inWi, inW1, inW2, inWo, 128, 256, 128, XT, nullptr, BF_H);

  run_gat(4, BF_H, gWs[0], dBs[0], gWd[0], dBd[0], dAt[0], F_SK0, BF_H);  // down[0]
  run_gat(4, BF_H, gWs[1], dBs[1], gWd[1], dBd[1], dAt[1], F_SK1, BF_H);  // down[1]

  run_mlp(mMid, miWi, miW1, miW2, miWo, 128, 128, 128, BF_H, F_HC, nullptr);

  int stot4 = N * 32;
  sum_to_bf16_k<<<(stot4 + 255) / 256, 256, 0, stream>>>(F_HC, F_SK1, BF_H, stot4);  // h += down[1]
  run_gat(4, BF_H, gWs[2], uBs[0], gWd[2], uBd[0], uAt[0], F_HC, nullptr);           // up[0]
  sum_to_bf16_k<<<(stot4 + 255) / 256, 256, 0, stream>>>(F_HC, F_SK0, BF_H, stot4);  // h += down[0]
  run_gat(1, BF_H, gWs[3], uBs[1], gWd[3], uBd[1], uAt[1], nullptr, BF_H);           // up[1], 1 head

  run_mlp(mOut, moWi, moW1, moW2, moWo, 128, 128, 128, BF_H, F_HC, nullptr);

  hipMemsetAsync(d_out, 0, sizeof(float), stream);
  loss_k<<<(N + 7) / 8, 256, 0, stream>>>(F_HC, y, (float*)d_out, N);
}
